// MyGPT_73186242724420
// MI455X (gfx1250) — compile-verified, matrix operand order checked by reading, not run on hardware
//
#include <hip/hip_runtime.h>
#include <hip/hip_bf16.h>

// ---------------------------------------------------------------------------
// Types for CDNA5 WMMA (wave32, 16x16x32 bf16 -> f32)
// ---------------------------------------------------------------------------
typedef __attribute__((ext_vector_type(16))) __bf16 bf16x16;
typedef __attribute__((ext_vector_type(8)))  float  f32x8;
typedef __attribute__((ext_vector_type(4)))  int    i32x4;

union AFrag { bf16x16 v; unsigned int u[8]; };
union BFrag { bf16x16 v; unsigned int u[8]; };

__device__ __forceinline__ unsigned short f2bf(float f) {
  union { float f; unsigned int u; } x; x.f = f;
  unsigned int u = x.u;
  unsigned int r = (u + 0x7FFFu + ((u >> 16) & 1u)) >> 16;
  return (unsigned short)r;
}

// ---- gfx1250 async global->LDS staging (ASYNCcnt-tracked DMA path) --------
#if defined(__HIP_DEVICE_COMPILE__) && __has_builtin(__builtin_amdgcn_global_load_async_to_lds_b128)
#define USE_ASYNC_LDS 1
#else
#define USE_ASYNC_LDS 0
#endif

#if USE_ASYNC_LDS
// builtin signature (from clang diagnostic): (int4 AS1*, int4 AS3*, imm, imm)
__device__ __forceinline__ __attribute__((address_space(1))) i32x4*
as_g128(const unsigned short* p) {
  return (__attribute__((address_space(1))) i32x4*)(unsigned long long)p;
}
__device__ __forceinline__ __attribute__((address_space(3))) i32x4*
as_l128(unsigned short* p) {
  // generic LDS pointers carry the LDS byte offset in addr[31:0] (ISA 10.2)
  return (__attribute__((address_space(3))) i32x4*)(unsigned int)(unsigned long long)p;
}
#endif

__device__ __forceinline__ void wait_async0() {
#if __has_builtin(__builtin_amdgcn_s_wait_asynccnt)
  __builtin_amdgcn_s_wait_asynccnt(0);
#else
  asm volatile("s_wait_asynccnt 0x0" ::: "memory");
#endif
}

// ---------------------------------------------------------------------------
// Batched WMMA GEMM:  C[z] = act( A[z] @ B[z]^T + bias + residual )
//   A: bf16 [M,K] row-major (lda);  B: bf16 [N,K] row-major (ldb)
//   C: f32 or bf16 [M,N] (ldc); TRANSC stores C[col*ldc + row] instead.
//   Batch offset: (z/div)*s0 + (z%div)*s1 per operand.
// REQUIREMENT (all call sites satisfy it): K % 32 == 0, lda/ldb % 8 == 0,
// operand base offsets 16B-aligned. M/N raggedness handled by clamped staging
// (reads stay in-bounds; epilogue bounds-checks stores).
// Block tile 128x128, K-step 32, 256 threads = 8 waves (4 in M x 2 in N),
// double-buffered LDS, 8 v_wmma_f32_16x16x32_bf16 per wave per K-step.
// ---------------------------------------------------------------------------
struct GemmP {
  const unsigned short* A;
  const unsigned short* B;
  void* C;
  const float* bias;   // length N, may be null
  const float* res;    // same layout/offsets as C (f32), may be null
  int M, N, K, lda, ldb, ldc;
  long long sA0, sA1; int divA;
  long long sB0, sB1; int divB;
  long long sC0, sC1; int divC;
  float alpha;
};

#define TM 128
#define TN 128
#define TK 32
#define TKP 40  // padded LDS stride in bf16 elems (80B rows: 16B-aligned, conflict-free)

template<bool OUTBF16, bool RELU, bool TRANSC>
__global__ __launch_bounds__(256)
void gemm_wmma(GemmP p) {
  __shared__ __align__(16) unsigned short As[2][TM * TKP];
  __shared__ __align__(16) unsigned short Bs[2][TN * TKP];

  const int tid  = threadIdx.x;
  const int lane = tid & 31;
  const int wave = tid >> 5;
  const int wm = wave & 3;          // 0..3  (M direction, 32 rows each)
  const int wn = wave >> 2;         // 0..1  (N direction, 64 cols each)
  const int laneHalf = lane >> 4;   // 0/1
  const int lane15   = lane & 15;

  const int m0 = blockIdx.y * TM;
  const int n0 = blockIdx.x * TN;
  const int z  = blockIdx.z;

  const long long offA = (long long)(z / p.divA) * p.sA0 + (long long)(z % p.divA) * p.sA1;
  const long long offB = (long long)(z / p.divB) * p.sB0 + (long long)(z % p.divB) * p.sB1;
  const long long offC = (long long)(z / p.divC) * p.sC0 + (long long)(z % p.divC) * p.sC1;

  // ---- clamped, unguarded 128-bit staging (per-thread: one A row + one B row)
  const int r  = tid >> 1;
  const int kb = (tid & 1) * 16;
  const int ra = min(m0 + r, p.M - 1);
  const int rb = min(n0 + r, p.N - 1);
  const unsigned short* gpa = p.A + offA + (long long)ra * p.lda + kb;
  const unsigned short* gpb = p.B + offB + (long long)rb * p.ldb + kb;
  unsigned short* lpa0 = &As[0][r * TKP + kb];
  unsigned short* lpb0 = &Bs[0][r * TKP + kb];
  unsigned short* lpa1 = &As[1][r * TKP + kb];
  unsigned short* lpb1 = &Bs[1][r * TKP + kb];

  auto stage = [&](int buf, int kt) {
    unsigned short* la = buf ? lpa1 : lpa0;
    unsigned short* lb = buf ? lpb1 : lpb0;
#if USE_ASYNC_LDS
    __builtin_amdgcn_global_load_async_to_lds_b128(as_g128(gpa + kt),     as_l128(la),     0, 0);
    __builtin_amdgcn_global_load_async_to_lds_b128(as_g128(gpa + kt + 8), as_l128(la + 8), 0, 0);
    __builtin_amdgcn_global_load_async_to_lds_b128(as_g128(gpb + kt),     as_l128(lb),     0, 0);
    __builtin_amdgcn_global_load_async_to_lds_b128(as_g128(gpb + kt + 8), as_l128(lb + 8), 0, 0);
#else
    uint4 a0 = *(const uint4*)(gpa + kt);
    uint4 a1 = *(const uint4*)(gpa + kt + 8);
    uint4 b0 = *(const uint4*)(gpb + kt);
    uint4 b1 = *(const uint4*)(gpb + kt + 8);
    *(uint4*)la       = a0;
    *(uint4*)(la + 8) = a1;
    *(uint4*)lb       = b0;
    *(uint4*)(lb + 8) = b1;
#endif
  };

  f32x8 acc[2][4];
  #pragma unroll
  for (int i = 0; i < 2; ++i)
    #pragma unroll
    for (int j = 0; j < 4; ++j)
      acc[i][j] = {};

  auto compute_tile = [&](int buf) {
    BFrag bfr[4];
    const int base = laneHalf * 8;
    #pragma unroll
    for (int j = 0; j < 4; ++j) {
      const unsigned short* bp = &Bs[buf][(wn * 64 + j * 16 + lane15) * TKP];
      #pragma unroll
      for (int q = 0; q < 4; ++q) {
        bfr[j].u[q]     = *(const unsigned int*)&bp[base + 2 * q];
        bfr[j].u[4 + q] = *(const unsigned int*)&bp[16 + base + 2 * q];
      }
    }
    #pragma unroll
    for (int i = 0; i < 2; ++i) {
      AFrag af;
      const unsigned short* ap = &As[buf][(wm * 32 + i * 16 + lane15) * TKP];
      #pragma unroll
      for (int q = 0; q < 4; ++q) {
        af.u[q]     = *(const unsigned int*)&ap[base + 2 * q];
        af.u[4 + q] = *(const unsigned int*)&ap[16 + base + 2 * q];
      }
      #pragma unroll
      for (int j = 0; j < 4; ++j) {
        acc[i][j] = __builtin_amdgcn_wmma_f32_16x16x32_bf16(
            false, af.v, false, bfr[j].v, (short)0, acc[i][j], false, false);
      }
    }
  };

  // ---- main loop: double-buffered, one barrier per K-step
  // Per iter: wait own async tile -> barrier (tile visible everywhere,
  // previous-buffer readers drained) -> issue next tile's DMA -> WMMA.
  stage(0, 0);
  int cur = 0;
  for (int kt = 0; kt < p.K; kt += TK) {
#if USE_ASYNC_LDS
    wait_async0();
#endif
    __syncthreads();
    if (kt + TK < p.K) {
      stage(cur ^ 1, kt + TK);
      if (kt + 2 * TK < p.K) {
        __builtin_prefetch(gpa + kt + 2 * TK, 0, 1);
        __builtin_prefetch(gpb + kt + 2 * TK, 0, 1);
      }
    }
    compute_tile(cur);
    cur ^= 1;
  }

  // ---- epilogue: alpha, bias, residual, relu, store f32/bf16 (bounds-checked)
  #pragma unroll
  for (int i = 0; i < 2; ++i) {
    #pragma unroll
    for (int j = 0; j < 4; ++j) {
      const int col = n0 + wn * 64 + j * 16 + lane15;
      if (col >= p.N) continue;
      const float bv = (p.bias != nullptr) ? p.bias[col] : 0.0f;
      #pragma unroll
      for (int rr = 0; rr < 8; ++rr) {
        const int row = m0 + wm * 32 + i * 16 + laneHalf * 8 + rr;
        if (row >= p.M) continue;
        float v = acc[i][j][rr] * p.alpha + bv;
        const long long ci = TRANSC
            ? offC + (long long)col * p.ldc + row
            : offC + (long long)row * p.ldc + col;
        if (p.res != nullptr) v += p.res[ci];
        if (RELU) v = fmaxf(v, 0.0f);
        if (OUTBF16) ((unsigned short*)p.C)[ci] = f2bf(v);
        else         ((float*)p.C)[ci] = v;
      }
    }
  }
}

// ---------------------------------------------------------------------------
// Elementwise / reduction kernels
// ---------------------------------------------------------------------------
__global__ __launch_bounds__(256)
void embed_k(const int* __restrict__ ctx, const float* __restrict__ tok,
             const float* __restrict__ pos, float* __restrict__ x,
             int Ss, int Dm) {
  const long long row = blockIdx.x;
  const int s = (int)(row % Ss);
  const int t = ctx[row];
  const float* tr = tok + (long long)t * Dm;
  const float* pr = pos + (long long)s * Dm;
  float* xr = x + row * Dm;
  for (int c = threadIdx.x; c < Dm; c += 256) xr[c] = tr[c] + pr[c];
}

__global__ __launch_bounds__(256)
void layernorm_k(const float* __restrict__ x, const float* __restrict__ g,
                 const float* __restrict__ b, unsigned short* __restrict__ out,
                 int Dm) {
  const long long row = blockIdx.x;
  const float* xr = x + row * Dm;
  unsigned short* orow = out + row * Dm;
  const int tid = threadIdx.x;

  float s = 0.f, ss = 0.f;
  for (int c = tid; c < Dm; c += 256) { float v = xr[c]; s += v; ss += v * v; }
  #pragma unroll
  for (int o = 16; o > 0; o >>= 1) { s += __shfl_xor(s, o, 32); ss += __shfl_xor(ss, o, 32); }
  __shared__ float rs[8], rss[8];
  if ((tid & 31) == 0) { rs[tid >> 5] = s; rss[tid >> 5] = ss; }
  __syncthreads();
  s = 0.f; ss = 0.f;
  #pragma unroll
  for (int w = 0; w < 8; ++w) { s += rs[w]; ss += rss[w]; }
  const float mean = s / Dm;
  const float var  = ss / Dm - mean * mean;
  const float inv  = rsqrtf(var + 1e-5f);
  for (int c = tid; c < Dm; c += 256)
    orow[c] = f2bf((xr[c] - mean) * inv * g[c] + b[c]);
}

__global__ __launch_bounds__(256)
void softmax_causal_k(const float* __restrict__ scores,
                      unsigned short* __restrict__ attn, int Ss, float scale) {
  const long long row = blockIdx.x;            // (b*H + h)*S + q
  const int q = (int)(row % Ss);
  const float* src = scores + row * Ss;
  unsigned short* dst = attn + row * Ss;
  const int tid = threadIdx.x;

  __shared__ float buf[1024];
  __shared__ float red[8];

  float lmax = -3.4e38f;
  for (int c = tid; c < Ss; c += 256) {
    float v = (c <= q) ? src[c] * scale : -3.4e38f;
    buf[c] = v;
    lmax = fmaxf(lmax, v);
  }
  #pragma unroll
  for (int o = 16; o > 0; o >>= 1) lmax = fmaxf(lmax, __shfl_xor(lmax, o, 32));
  if ((tid & 31) == 0) red[tid >> 5] = lmax;
  __syncthreads();
  float m = red[0];
  #pragma unroll
  for (int w = 1; w < 8; ++w) m = fmaxf(m, red[w]);

  float lsum = 0.f;
  for (int c = tid; c < Ss; c += 256) {
    float e = __expf(buf[c] - m);
    buf[c] = e;
    lsum += e;
  }
  #pragma unroll
  for (int o = 16; o > 0; o >>= 1) lsum += __shfl_xor(lsum, o, 32);
  __syncthreads();
  if ((tid & 31) == 0) red[tid >> 5] = lsum;
  __syncthreads();
  float sum = 0.f;
  #pragma unroll
  for (int w = 0; w < 8; ++w) sum += red[w];
  const float inv = 1.0f / sum;
  for (int c = tid; c < Ss; c += 256) dst[c] = f2bf(buf[c] * inv);
}

// Transpose-convert one stack of L matrices: src f32 [L,K,N] -> dst bf16 [L,N,K]
__global__ __launch_bounds__(256)
void cvtT_k(const float* __restrict__ src, unsigned short* __restrict__ dst,
            int K, int N) {
  __shared__ float t[32][33];
  const int l = blockIdx.z;
  const float* s = src + (long long)l * K * N;
  unsigned short* d = dst + (long long)l * N * K;
  const int k0 = blockIdx.y * 32, n0 = blockIdx.x * 32;
  const int tx = threadIdx.x & 31, ty = threadIdx.x >> 5;  // 32 x 8
  #pragma unroll
  for (int i = 0; i < 32; i += 8) {
    const int k = k0 + ty + i, n = n0 + tx;
    t[ty + i][tx] = (k < K && n < N) ? s[(long long)k * N + n] : 0.f;
  }
  __syncthreads();
  #pragma unroll
  for (int i = 0; i < 32; i += 8) {
    const int n = n0 + ty + i, k = k0 + tx;
    if (n < N && k < K) d[(long long)n * K + k] = f2bf(t[tx][ty + i]);
  }
}

// ---------------------------------------------------------------------------
// Host-side orchestration
// ---------------------------------------------------------------------------
static inline void launch_gemm(const GemmP& p, int batches, bool outbf16,
                               bool relu, bool transc, hipStream_t st) {
  dim3 grid((p.N + TN - 1) / TN, (p.M + TM - 1) / TM, batches);
  dim3 block(256);
  if (outbf16 && !relu && !transc)      gemm_wmma<true,  false, false><<<grid, block, 0, st>>>(p);
  else if (outbf16 && !relu && transc)  gemm_wmma<true,  false, true ><<<grid, block, 0, st>>>(p);
  else if (outbf16 && relu)             gemm_wmma<true,  true,  false><<<grid, block, 0, st>>>(p);
  else                                  gemm_wmma<false, false, false><<<grid, block, 0, st>>>(p);
}

static inline GemmP mkgemm(const unsigned short* A, const unsigned short* B,
                           void* C, int M, int N, int K, int lda, int ldb, int ldc) {
  GemmP p{};
  p.A = A; p.B = B; p.C = C;
  p.bias = nullptr; p.res = nullptr;
  p.M = M; p.N = N; p.K = K; p.lda = lda; p.ldb = ldb; p.ldc = ldc;
  p.sA0 = p.sA1 = p.sB0 = p.sB1 = p.sC0 = p.sC1 = 0;
  p.divA = p.divB = p.divC = 1;
  p.alpha = 1.0f;
  return p;
}

extern "C" void kernel_launch(void* const* d_in, const int* in_sizes, int n_in,
                              void* d_out, int out_size, void* d_ws, size_t ws_size,
                              hipStream_t stream) {
  (void)in_sizes; (void)n_in; (void)out_size; (void)ws_size;

  const int Vv = 32000, Dm = 768, Hn = 12, Ln = 6, Bb = 2, Ss = 1024;
  const int HDd = Dm / Hn;              // 64
  const int Ff = 4 * Dm;                // 3072
  const int Mr = Bb * Ss;               // 2048 rows of the token stream
  const long long DD = (long long)Dm * Dm;

  const int*   ctx   = (const int*)d_in[0];
  const float* tok   = (const float*)d_in[1];
  const float* pos   = (const float*)d_in[2];
  const float* Wq    = (const float*)d_in[3];
  const float* Wk    = (const float*)d_in[4];
  const float* Wv    = (const float*)d_in[5];
  const float* Wo    = (const float*)d_in[6];
  const float* bo    = (const float*)d_in[7];
  const float* ln1g  = (const float*)d_in[8];
  const float* ln1b  = (const float*)d_in[9];
  const float* W1    = (const float*)d_in[10];
  const float* b1    = (const float*)d_in[11];
  const float* W2    = (const float*)d_in[12];
  const float* b2    = (const float*)d_in[13];
  const float* ln2g  = (const float*)d_in[14];
  const float* ln2b  = (const float*)d_in[15];
  const float* lnfg  = (const float*)d_in[16];
  const float* lnfb  = (const float*)d_in[17];
  const float* Wout  = (const float*)d_in[18];
  const float* bout  = (const float*)d_in[19];
  float* out = (float*)d_out;

  // ---- workspace bump allocator
  char* wsb = (char*)d_ws;
  size_t off = 0;
  auto alloc = [&](size_t bytes) -> void* {
    void* p = wsb + off;
    off = (off + bytes + 255) & ~(size_t)255;
    return p;
  };

  // weights stored TRANSPOSED (bf16 [N,K]) for the K-contiguous GEMM path
  unsigned short* wq_bf  = (unsigned short*)alloc((size_t)Ln * DD * 2);
  unsigned short* wk_bf  = (unsigned short*)alloc((size_t)Ln * DD * 2);
  unsigned short* wv_bf  = (unsigned short*)alloc((size_t)Ln * DD * 2);
  unsigned short* wo_bf  = (unsigned short*)alloc((size_t)Ln * DD * 2);
  unsigned short* w1_bf  = (unsigned short*)alloc((size_t)Ln * Dm * Ff * 2);  // [Ff,Dm] per layer
  unsigned short* w2_bf  = (unsigned short*)alloc((size_t)Ln * Ff * Dm * 2);  // [Dm,Ff] per layer
  unsigned short* wout_bf= (unsigned short*)alloc((size_t)Dm * Vv * 2);       // [Vv,Dm]
  float*          x      = (float*)alloc((size_t)Mr * Dm * 4);
  unsigned short* h_bf   = (unsigned short*)alloc((size_t)Mr * Dm * 2);
  unsigned short* q_bf   = (unsigned short*)alloc((size_t)Mr * Dm * 2);
  unsigned short* k_bf   = (unsigned short*)alloc((size_t)Mr * Dm * 2);
  unsigned short* vt_bf  = (unsigned short*)alloc((size_t)Bb * Dm * Ss * 2);  // V^T: [B][D][S]
  float*          scores = (float*)alloc((size_t)Bb * Hn * Ss * Ss * 4);
  unsigned short* attn_bf= (unsigned short*)alloc((size_t)Bb * Hn * Ss * Ss * 2);
  unsigned short* ctx_bf = (unsigned short*)alloc((size_t)Mr * Dm * 2);
  unsigned short* ffn_bf = (unsigned short*)alloc((size_t)Mr * Ff * 2);

  // ---- weight transpose+conversion f32 [K,N] -> bf16 [N,K]
  auto cvtT = [&](const float* src, unsigned short* dst, int K, int N, int L_) {
    dim3 grid((N + 31) / 32, (K + 31) / 32, L_);
    cvtT_k<<<grid, 256, 0, stream>>>(src, dst, K, N);
  };
  cvtT(Wq,   wq_bf,  Dm, Dm, Ln);
  cvtT(Wk,   wk_bf,  Dm, Dm, Ln);
  cvtT(Wv,   wv_bf,  Dm, Dm, Ln);
  cvtT(Wo,   wo_bf,  Dm, Dm, Ln);
  cvtT(W1,   w1_bf,  Dm, Ff, Ln);
  cvtT(W2,   w2_bf,  Ff, Dm, Ln);
  cvtT(Wout, wout_bf,Dm, Vv, 1);

  // ---- embedding: x = tok_emb[context] + pos_emb
  embed_k<<<Mr, 256, 0, stream>>>(ctx, tok, pos, x, Ss, Dm);

  const float attn_scale = 0.125f;  // 1/sqrt(64)

  for (int l = 0; l < Ln; ++l) {
    // LN1 -> h_bf
    layernorm_k<<<Mr, 256, 0, stream>>>(x, ln1g + (long long)l * Dm,
                                        ln1b + (long long)l * Dm, h_bf, Dm);
    // Q, K projections (weights transposed: [Dm_out, Dm_in], ldb = Dm)
    {
      GemmP p = mkgemm(h_bf, wq_bf + (long long)l * DD, q_bf, Mr, Dm, Dm, Dm, Dm, Dm);
      launch_gemm(p, 1, true, false, false, stream);
      p = mkgemm(h_bf, wk_bf + (long long)l * DD, k_bf, Mr, Dm, Dm, Dm, Dm, Dm);
      launch_gemm(p, 1, true, false, false, stream);
    }
    // V projection, stored transposed: vt[b][d][s]  (batched over b, TRANSC)
    {
      GemmP p = mkgemm(h_bf, wv_bf + (long long)l * DD, vt_bf, Ss, Dm, Dm, Dm, Dm, Ss);
      p.sA0 = (long long)Ss * Dm;  p.divA = 1;
      p.sC0 = (long long)Dm * Ss;  p.divC = 1;
      launch_gemm(p, Bb, true, false, true, stream);
    }
    // scores[b,h] = Q[b,:,h,:] @ K[b,:,h,:]^T   (batched over z = b*H + h)
    {
      GemmP p = mkgemm(q_bf, k_bf, scores, Ss, Ss, HDd, Dm, Dm, Ss);
      p.sA0 = (long long)Ss * Dm; p.sA1 = HDd; p.divA = Hn;
      p.sB0 = (long long)Ss * Dm; p.sB1 = HDd; p.divB = Hn;
      p.sC0 = (long long)Hn * Ss * Ss; p.sC1 = (long long)Ss * Ss; p.divC = Hn;
      launch_gemm(p, Bb * Hn, false, false, false, stream);
    }
    // causal softmax with scale -> bf16 attn
    softmax_causal_k<<<Bb * Hn * Ss, 256, 0, stream>>>(scores, attn_bf, Ss, attn_scale);
    // ctx[b,:,h,:] = attn[b,h] @ vt[b, h*HD:(h+1)*HD, :]^T -> merged [B,S,D] bf16
    {
      GemmP p = mkgemm(attn_bf, vt_bf, ctx_bf, Ss, HDd, Ss, Ss, Ss, Dm);
      p.sA0 = (long long)Hn * Ss * Ss; p.sA1 = (long long)Ss * Ss; p.divA = Hn;
      p.sB0 = (long long)Dm * Ss; p.sB1 = (long long)HDd * Ss; p.divB = Hn;
      p.sC0 = (long long)Ss * Dm; p.sC1 = HDd; p.divC = Hn;
      launch_gemm(p, Bb * Hn, true, false, false, stream);
    }
    // x = x + ctx @ Wo + bo
    {
      GemmP p = mkgemm(ctx_bf, wo_bf + (long long)l * DD, x, Mr, Dm, Dm, Dm, Dm, Dm);
      p.bias = bo + (long long)l * Dm;
      p.res  = x;
      launch_gemm(p, 1, false, false, false, stream);
    }
    // LN2 -> h_bf
    layernorm_k<<<Mr, 256, 0, stream>>>(x, ln2g + (long long)l * Dm,
                                        ln2b + (long long)l * Dm, h_bf, Dm);
    // FFN1: relu(h @ W1 + b1) -> bf16  (W1^T is [Ff, Dm], ldb = Dm)
    {
      GemmP p = mkgemm(h_bf, w1_bf + (long long)l * Dm * Ff, ffn_bf, Mr, Ff, Dm, Dm, Dm, Ff);
      p.bias = b1 + (long long)l * Ff;
      launch_gemm(p, 1, true, true, false, stream);
    }
    // FFN2: x = x + ffn @ W2 + b2  (W2^T is [Dm, Ff], ldb = Ff)
    {
      GemmP p = mkgemm(ffn_bf, w2_bf + (long long)l * Ff * Dm, x, Mr, Dm, Ff, Ff, Ff, Dm);
      p.bias = b2 + (long long)l * Dm;
      p.res  = x;
      launch_gemm(p, 1, false, false, false, stream);
    }
  }

  // final LN -> h_bf, then logits = h @ Wout + bout  (Wout^T is [Vv, Dm], ldb = Dm)
  layernorm_k<<<Mr, 256, 0, stream>>>(x, lnfg, lnfb, h_bf, Dm);
  {
    GemmP p = mkgemm(h_bf, wout_bf, out, Mr, Vv, Dm, Dm, Dm, Vv);
    p.bias = bout;
    launch_gemm(p, 1, false, false, false, stream);
  }
}
